// Model_89103391523683
// MI455X (gfx1250) — matrix operand order repaired and checked by reading, not compiled, not run on hardware
//
#include <hip/hip_runtime.h>
#include <hip/hip_bf16.h>

#define N_MQ 50000
#define N_SQ 50000
#define NE   600000
#define EL   200000
#define H    128
#define F_IN 768
#define N_LAYERS 4
#define OFFN 50016   // 50001 padded to multiple of 16 ints (keeps 32B alignment)

typedef __bf16 bf16_t;
typedef __attribute__((ext_vector_type(16))) bf16_t v16bf;
typedef __attribute__((ext_vector_type(8)))  float  v8f;

// ---------------------------------------------------------------------------
// A-fragment loader for v_wmma_f32_16x16x32_bf16 (wave32), converting f32->bf16.
// A: 16x32 (MxK): lane L(0-15) row M=L holds K={0..7,16..23},
//                 lane L+16 row M=L holds K={8..15,24..31}.
__device__ __forceinline__ v16bf load_a_frag(const float* __restrict__ base,
                                             int ld, int row, int kb, int lane) {
  const float* p0 = base + (long long)row * ld + kb + 8 * (lane >> 4);
  const float* p1 = p0 + 16;
  v16bf a;
#pragma unroll
  for (int j = 0; j < 8; ++j) a[j] = (bf16_t)p0[j];
#pragma unroll
  for (int j = 0; j < 8; ++j) a[8 + j] = (bf16_t)p1[j];
  return a;
}

// Packed-B fragment: 16 bf16 contiguous per lane (32B aligned) — one wide load.
__device__ __forceinline__ v16bf load_b_packed(const bf16_t* __restrict__ P,
                                               int frag, int lane) {
  return *(const v16bf*)(P + (((long long)frag * 32 + lane) << 4));
}

// ---------------------------------------------------------------------------
// Pack W [Hout x K] f32 row-major into B^T fragment-linear bf16.
__global__ void k_pack_w(const float* __restrict__ W, bf16_t* __restrict__ out,
                         int K, int Hout) {
  int idx = blockIdx.x * blockDim.x + threadIdx.x;
  int total = K * Hout;
  if (idx >= total) return;
  const int j    = idx & 15;
  const int lane = (idx >> 4) & 31;
  const int frag = idx >> 9;
  const int ntiles = Hout >> 4;
  const int kt = frag / ntiles, t = frag % ntiles;
  const int k = kt * 32 + 8 * (lane >> 4) + (j & 7) + 16 * (j >> 3);
  const int n = t * 16 + (lane & 15);
  out[idx] = (bf16_t)W[(long long)n * K + k];
}

// ---------------------------------------------------------------------------
// Encoder: xsq = sq_x @ lin_W^T + lin_b + movie_emb   (M=N_SQ, K=F_IN, N=H)
__global__ __launch_bounds__(256) void k_encode_sq(
    const float* __restrict__ sq_x, const bf16_t* __restrict__ packW,
    const float* __restrict__ lin_b, const float* __restrict__ movie_emb,
    float* __restrict__ xsq, int nRows) {
  const int lane = threadIdx.x & 31;
  const int wave = threadIdx.x >> 5;
  const int rowBase = (blockIdx.x * (blockDim.x >> 5) + wave) * 16;
  if (rowBase >= nRows) return;  // wave-uniform; EXEC all-ones inside WMMA

  v8f acc[8];
#pragma unroll
  for (int t = 0; t < 8; ++t)
#pragma unroll
    for (int r = 0; r < 8; ++r) acc[t][r] = 0.0f;

  const int m = lane & 15;
  for (int kb = 0; kb < F_IN; kb += 32) {
    const int kt = kb >> 5;
    v16bf a = load_a_frag(sq_x, F_IN, rowBase + m, kb, lane);
#pragma unroll
    for (int t = 0; t < 8; ++t) {
      v16bf b = load_b_packed(packW, kt * 8 + t, lane);
      acc[t] = __builtin_amdgcn_wmma_f32_16x16x32_bf16(
          false, a, false, b, (short)0, acc[t], false, false);
    }
  }
  const int n0 = lane & 15;
  const int rOff = (lane >> 4) * 8;
#pragma unroll
  for (int t = 0; t < 8; ++t) {
    const int n = t * 16 + n0;
    const float bv = lin_b[n];
#pragma unroll
    for (int r = 0; r < 8; ++r) {
      const int row = rowBase + rOff + r;
      xsq[(long long)row * H + n] = acc[t][r] + bv + movie_emb[(long long)row * H + n];
    }
  }
}

// ---------------------------------------------------------------------------
// SAGE update: out = mean @ Wl^T + bl + xold @ Wr^T, optional ReLU.  K=N=H=128.
// 5 waves/block * 625 blocks == 3125 row-tiles exactly (no divergent barriers).
#define SAGE_WAVES 5
#define SAGE_TPB   (SAGE_WAVES * 32)
#define WFRAG_ELEMS ((H / 32) * (H / 16) * 32 * 16)  // 16384 bf16 = 32 KB

__global__ __launch_bounds__(SAGE_TPB) void k_sage_update(
    const float* __restrict__ mean, const float* __restrict__ xold,
    const bf16_t* __restrict__ Pl, const float* __restrict__ bl,
    const bf16_t* __restrict__ Pr, float* __restrict__ out, int relu) {
  __shared__ __attribute__((aligned(32))) bf16_t ldsW[WFRAG_ELEMS];
  const int lane = threadIdx.x & 31;
  const int wave = threadIdx.x >> 5;
  const int rowBase = (blockIdx.x * SAGE_WAVES + wave) * 16;

  v8f acc[8];
#pragma unroll
  for (int t = 0; t < 8; ++t)
#pragma unroll
    for (int r = 0; r < 8; ++r) acc[t][r] = 0.0f;

  const int m = lane & 15;

  // ---- phase 1: mean @ Wl^T ------------------------------------------------
  for (int i = threadIdx.x; i < WFRAG_ELEMS / 8; i += SAGE_TPB)
    ((uint4*)ldsW)[i] = ((const uint4*)Pl)[i];
  __syncthreads();
#pragma unroll
  for (int kb = 0; kb < H; kb += 32) {
    const int kt = kb >> 5;
    v16bf a = load_a_frag(mean, H, rowBase + m, kb, lane);
#pragma unroll
    for (int t = 0; t < 8; ++t) {
      v16bf b = *(const v16bf*)(ldsW + (((kt * 8 + t) * 32 + lane) << 4));
      acc[t] = __builtin_amdgcn_wmma_f32_16x16x32_bf16(
          false, a, false, b, (short)0, acc[t], false, false);
    }
  }
  __syncthreads();

  // ---- phase 2: + xold @ Wr^T ----------------------------------------------
  for (int i = threadIdx.x; i < WFRAG_ELEMS / 8; i += SAGE_TPB)
    ((uint4*)ldsW)[i] = ((const uint4*)Pr)[i];
  __syncthreads();
#pragma unroll
  for (int kb = 0; kb < H; kb += 32) {
    const int kt = kb >> 5;
    v16bf a = load_a_frag(xold, H, rowBase + m, kb, lane);
#pragma unroll
    for (int t = 0; t < 8; ++t) {
      v16bf b = *(const v16bf*)(ldsW + (((kt * 8 + t) * 32 + lane) << 4));
      acc[t] = __builtin_amdgcn_wmma_f32_16x16x32_bf16(
          false, a, false, b, (short)0, acc[t], false, false);
    }
  }

  // ---- epilogue --------------------------------------------------------------
  const int n0 = lane & 15;
  const int rOff = (lane >> 4) * 8;
#pragma unroll
  for (int t = 0; t < 8; ++t) {
    const int n = t * 16 + n0;
    const float bv = bl[n];
#pragma unroll
    for (int r = 0; r < 8; ++r) {
      const int row = rowBase + rOff + r;
      float v = acc[t][r] + bv;
      if (relu) v = fmaxf(v, 0.0f);
      out[(long long)row * H + n] = v;
    }
  }
}

// ---------------------------------------------------------------------------
// CSR build + gather-mean aggregation (replaces 614M f32 atomics with
// coalesced streaming reads; only int atomics remain in the build).
__global__ void k_zero_i32(int* __restrict__ p, int n) {
  int i = blockIdx.x * blockDim.x + threadIdx.x;
  if (i < n) p[i] = 0;
}
__global__ void k_deg(const int* __restrict__ src, const int* __restrict__ dst,
                      int* __restrict__ deg_mq, int* __restrict__ deg_sq, int nE) {
  int e = blockIdx.x * blockDim.x + threadIdx.x;
  if (e < nE) {
    atomicAdd(&deg_sq[dst[e]], 1);
    atomicAdd(&deg_mq[src[e]], 1);
  }
}
// single-block chunked Hillis-Steele exclusive scan (n ~ 50k: trivial work)
#define SCAN_TPB 1024
__global__ __launch_bounds__(SCAN_TPB) void k_scan_excl(
    const int* __restrict__ deg, int* __restrict__ off, int n) {
  __shared__ int buf[SCAN_TPB];
  __shared__ int carry;
  if (threadIdx.x == 0) { carry = 0; off[0] = 0; }
  __syncthreads();
  for (int base = 0; base < n; base += SCAN_TPB) {
    const int i = base + threadIdx.x;
    buf[threadIdx.x] = (i < n) ? deg[i] : 0;
    __syncthreads();
    for (int o = 1; o < SCAN_TPB; o <<= 1) {
      int t = (threadIdx.x >= o) ? buf[threadIdx.x - o] : 0;
      __syncthreads();
      buf[threadIdx.x] += t;
      __syncthreads();
    }
    if (i < n) off[i + 1] = carry + buf[threadIdx.x];   // inclusive + carry
    __syncthreads();
    if (threadIdx.x == 0) carry += buf[SCAN_TPB - 1];
    __syncthreads();
  }
}
__global__ void k_fill_csr(const int* __restrict__ src, const int* __restrict__ dst,
                           int* __restrict__ cur_sq, int* __restrict__ cur_mq,
                           int* __restrict__ csr_sq, int* __restrict__ csr_mq, int nE) {
  int e = blockIdx.x * blockDim.x + threadIdx.x;
  if (e < nE) {
    const int s = src[e], d = dst[e];
    csr_sq[atomicAdd(&cur_sq[d], 1)] = s;   // in-neighbors (mq ids) of sq node d
    csr_mq[atomicAdd(&cur_mq[s], 1)] = d;   // in-neighbors (sq ids) of mq node s
  }
}
// one wave per destination node; lane owns 4 columns; coalesced 512B row reads
__global__ __launch_bounds__(256) void k_gather_mean(
    const float* __restrict__ xsrc, const int* __restrict__ csr,
    const int* __restrict__ off, float* __restrict__ mean, int nNodes) {
  const int lane = threadIdx.x & 31;
  const int wave = threadIdx.x >> 5;
  const int node = blockIdx.x * (blockDim.x >> 5) + wave;
  if (node >= nNodes) return;
  const int b = off[node], e = off[node + 1];   // wave-uniform
  const int c = lane * 4;
  float4 s = {0.f, 0.f, 0.f, 0.f};
  for (int i = b; i < e; ++i) {
    const int r = csr[i];                        // uniform -> scalar load
    const float4 v = *(const float4*)(xsrc + (long long)r * H + c);
    s.x += v.x; s.y += v.y; s.z += v.z; s.w += v.w;
  }
  const float sc = 1.0f / fmaxf((float)(e - b), 1.0f);
  float4 o = {s.x * sc, s.y * sc, s.z * sc, s.w * sc};
  *(float4*)(mean + (long long)node * H + c) = o;
}

// one wave per supervision edge: 128-dim dot, __shfl_xor reduce (wave32)
__global__ void k_classify(const float* __restrict__ xmq, const float* __restrict__ xsq,
                           const int* __restrict__ eli, float* __restrict__ out, int nE) {
  long long tid = (long long)blockIdx.x * blockDim.x + threadIdx.x;
  int e = (int)(tid >> 5);
  int lane = (int)tid & 31;
  if (e >= nE) return;
  const int a = eli[e];
  const int b = eli[nE + e];
  const float4 va = *(const float4*)(xmq + (long long)a * H + lane * 4);
  const float4 vb = *(const float4*)(xsq + (long long)b * H + lane * 4);
  float s = va.x * vb.x + va.y * vb.y + va.z * vb.z + va.w * vb.w;
#pragma unroll
  for (int off = 16; off; off >>= 1) s += __shfl_xor(s, off, 32);
  if (lane == 0) out[e] = s;
}

// ---------------------------------------------------------------------------
extern "C" void kernel_launch(void* const* d_in, const int* in_sizes, int n_in,
                              void* d_out, int out_size, void* d_ws, size_t ws_size,
                              hipStream_t stream) {
  (void)in_sizes; (void)n_in; (void)out_size; (void)ws_size;
  const float* sq_x      = (const float*)d_in[2];
  const int*   edge      = (const int*)d_in[3];   // [2, E]
  const int*   eli       = (const int*)d_in[4];   // [2, EL]
  const float* user_emb  = (const float*)d_in[5];
  const float* movie_emb = (const float*)d_in[6];
  const float* lin_W     = (const float*)d_in[7];
  const float* lin_b     = (const float*)d_in[8];
  const float* Wl_s      = (const float*)d_in[9];
  const float* bl_s      = (const float*)d_in[10];
  const float* Wr_s      = (const float*)d_in[11];
  const float* Wl_m      = (const float*)d_in[12];
  const float* bl_m      = (const float*)d_in[13];
  const float* Wr_m      = (const float*)d_in[14];
  float* out = (float*)d_out;

  const int* e_src = edge;        // mq index per edge
  const int* e_dst = edge + NE;   // sq index per edge

  // workspace carve-up: f32 region, then int region, then packed-bf16 region.
  float* ws = (float*)d_ws;
  const long long NHmq = (long long)N_MQ * H, NHsq = (long long)N_SQ * H;
  float* xmq[2]  = {ws, ws + NHmq};
  float* xsq[2]  = {ws + 2 * NHmq, ws + 2 * NHmq + NHsq};
  float* mean_sq = ws + 2 * NHmq + 2 * NHsq;
  float* mean_mq = mean_sq + NHsq;
  int* ip      = (int*)(mean_mq + NHmq);
  int* deg_sq  = ip;               ip += N_SQ;   // deg_sq|deg_mq contiguous
  int* deg_mq  = ip;               ip += N_MQ;
  int* off_sq  = ip;               ip += OFFN;
  int* off_mq  = ip;               ip += OFFN;
  int* cur_sq  = ip;               ip += N_SQ;
  int* cur_mq  = ip;               ip += N_MQ;
  int* csr_sq  = ip;               ip += NE;
  int* csr_mq  = ip;               ip += NE;
  bf16_t* packLin = (bf16_t*)ip;                       // F_IN*H packed bf16
  bf16_t* packLayer = packLin + (long long)F_IN * H;   // 16 matrices of H*H
  const long long PW = (long long)H * H;
  bf16_t* pWl_s = packLayer;
  bf16_t* pWr_s = pWl_s + 4 * PW;
  bf16_t* pWl_m = pWr_s + 4 * PW;
  bf16_t* pWr_m = pWl_m + 4 * PW;

  const int TPB = 256;

  // ---- pack all weights to bf16 fragment layout (deterministic each call) --
  k_pack_w<<<(F_IN * H + TPB - 1) / TPB, TPB, 0, stream>>>(lin_W, packLin, F_IN, H);
  for (int i = 0; i < N_LAYERS; ++i) {
    k_pack_w<<<(H * H + TPB - 1) / TPB, TPB, 0, stream>>>(Wl_s + i * H * H, pWl_s + i * PW, H, H);
    k_pack_w<<<(H * H + TPB - 1) / TPB, TPB, 0, stream>>>(Wr_s + i * H * H, pWr_s + i * PW, H, H);
    k_pack_w<<<(H * H + TPB - 1) / TPB, TPB, 0, stream>>>(Wl_m + i * H * H, pWl_m + i * PW, H, H);
    k_pack_w<<<(H * H + TPB - 1) / TPB, TPB, 0, stream>>>(Wr_m + i * H * H, pWr_m + i * PW, H, H);
  }

  // x_mq(0) = user_emb (mq_node_id is arange)
  hipMemcpyAsync(xmq[0], user_emb, NHmq * sizeof(float),
                 hipMemcpyDeviceToDevice, stream);
  // x_sq(0) = sq_x @ lin_W^T + lin_b + movie_emb  (sq_node_id is arange)
  const int encBlocks = (N_SQ / 16 + 7) / 8;
  k_encode_sq<<<encBlocks, TPB, 0, stream>>>(sq_x, packLin, lin_b, movie_emb,
                                             xsq[0], N_SQ);

  // ---- CSR build (graph is constant across layers) -------------------------
  k_zero_i32<<<(N_SQ + N_MQ + TPB - 1) / TPB, TPB, 0, stream>>>(deg_sq, N_SQ + N_MQ);
  k_deg<<<(NE + TPB - 1) / TPB, TPB, 0, stream>>>(e_src, e_dst, deg_mq, deg_sq, NE);
  k_scan_excl<<<1, SCAN_TPB, 0, stream>>>(deg_sq, off_sq, N_SQ);
  k_scan_excl<<<1, SCAN_TPB, 0, stream>>>(deg_mq, off_mq, N_MQ);
  hipMemcpyAsync(cur_sq, off_sq, N_SQ * sizeof(int), hipMemcpyDeviceToDevice, stream);
  hipMemcpyAsync(cur_mq, off_mq, N_MQ * sizeof(int), hipMemcpyDeviceToDevice, stream);
  k_fill_csr<<<(NE + TPB - 1) / TPB, TPB, 0, stream>>>(e_src, e_dst, cur_sq, cur_mq,
                                                       csr_sq, csr_mq, NE);

  const int gatherBlocksSq = (N_SQ + 7) / 8;   // 8 waves/block, wave per node
  const int gatherBlocksMq = (N_MQ + 7) / 8;
  const int sageBlocks = (N_SQ / 16) / SAGE_WAVES;  // 625: exact, barrier-safe

  for (int i = 0; i < N_LAYERS; ++i) {
    const int cur = i & 1, nxt = cur ^ 1;
    // mean_sq[d] = mean of x_mq over in-neighbors; mean_mq[s] = mean of x_sq
    k_gather_mean<<<gatherBlocksSq, TPB, 0, stream>>>(xmq[cur], csr_sq, off_sq,
                                                      mean_sq, N_SQ);
    k_gather_mean<<<gatherBlocksMq, TPB, 0, stream>>>(xsq[cur], csr_mq, off_mq,
                                                      mean_mq, N_MQ);
    // new_sq = mean_sq @ Wl_s^T + bl_s + x_sq @ Wr_s^T   (+ReLU on layer 0)
    k_sage_update<<<sageBlocks, SAGE_TPB, 0, stream>>>(
        mean_sq, xsq[cur], pWl_s + i * PW, bl_s + i * H, pWr_s + i * PW,
        xsq[nxt], i == 0 ? 1 : 0);
    // new_mq = mean_mq @ Wl_m^T + bl_m + x_mq @ Wr_m^T
    k_sage_update<<<sageBlocks, SAGE_TPB, 0, stream>>>(
        mean_mq, xmq[cur], pWl_m + i * PW, bl_m + i * H, pWr_m + i * PW,
        xmq[nxt], i == 0 ? 1 : 0);
  }
  // after 4 layers the final activations sit in buffer index 0
  const long long clsThreads = (long long)EL * 32;
  k_classify<<<(int)((clsThreads + TPB - 1) / TPB), TPB, 0, stream>>>(
      xmq[0], xsq[0], eli, out, EL);
}
